// Seq2seqLSTM_80676665688275
// MI455X (gfx1250) — matrix operand order repaired and checked by reading, not compiled, not run on hardware
//
#include <hip/hip_runtime.h>
#include <hip/hip_bf16.h>

// ---------------------------------------------------------------------------
// Seq2seq LSTM (encoder + attention decoder + vocab head + masked NLL)
// CDNA5 / gfx1250: bf16 WMMA (v_wmma_f32_16x16x32_bf16) for all GEMMs,
// fp32 accumulation, async GLOBAL->LDS A-tile staging (ASYNCcnt path),
// global_prefetch for L2-resident weights, 2-tile ILP per wave.
// ---------------------------------------------------------------------------

typedef __attribute__((ext_vector_type(16))) __bf16 v16bf;
typedef __attribute__((ext_vector_type(8)))  float  v8f;
typedef __attribute__((ext_vector_type(4)))  int    v4i;

struct BF16x16 { uint4 lo, hi; };   // 32B container for a v16bf fragment

__device__ __forceinline__ float sigmoidf(float x) { return 1.0f / (1.0f + __expf(-x)); }

#if defined(__HIP_DEVICE_COMPILE__) && \
    __has_builtin(__builtin_amdgcn_global_load_async_to_lds_b128) && \
    __has_builtin(__builtin_amdgcn_s_wait_asynccnt)
#define HAVE_ASYNC_LDS 1
// pointer-to-addrspace-qualified-int4 types expected by the async builtins
typedef __attribute__((address_space(1))) v4i* gas_v4i_ptr;   // global
typedef __attribute__((address_space(3))) v4i* las_v4i_ptr;   // LDS
#endif

// ---------------------------------------------------------------------------
// GEMM: C[M x N] (f32) = A_bf16[M x K] * W_bf16[N x K]^T + bias[N]
// grid.x = N/256, grid.y = M/16, block = 256 (8 waves)
// wave w owns N-tiles n0 = bx*256 + w*32 and n0+16.
// Dynamic LDS: 16*K bf16 (A row-block, shared by all 8 waves), filled with
// GLOBAL_LOAD_ASYNC_TO_LDS_B128 when available (ASYNCcnt), else uint4 copy.
// B fragment: row-major (N,K) weights give this code's contiguous-k B order:
//   lane l -> col n0+(l&15), 16 contiguous K at k0+(l>>4)*16  (one 32B load)
// A fragment (this code's contiguous k order, not the ISA table order):
//   lane l -> row (l&15), 16 contiguous K at k0+(l>>4)*16 (matches B)
// ---------------------------------------------------------------------------
__global__ void __launch_bounds__(256)
k_gemm_bf16(const __bf16* __restrict__ A, const __bf16* __restrict__ W,
            const float* __restrict__ bias, float* __restrict__ C,
            int N, int K)
{
    extern __shared__ __bf16 sA[];                 // 16 * K bf16
    const int tid   = threadIdx.x;
    const int wave  = tid >> 5;
    const int lane  = tid & 31;
    const int mrow  = lane & 15;
    const int khalf = lane >> 4;
    const int n0    = blockIdx.x * 256 + wave * 32;

    // ---- A row-block staging into LDS ------------------------------------
#ifdef HAVE_ASYNC_LDS
    {
        const int bytes = 16 * K * 2;              // A tile size in bytes
        const char* gsrc = (const char*)(A + (size_t)blockIdx.y * 16 * K);
        char*       ldst = (char*)sA;
        // each thread DMAs 16B per iteration; 256 threads -> 4KB per iter
        for (int o = tid * 16; o < bytes; o += 256 * 16) {
            __builtin_amdgcn_global_load_async_to_lds_b128(
                (gas_v4i_ptr)(gsrc + o),
                (las_v4i_ptr)(ldst + o),
                0, 0);
        }
        __builtin_amdgcn_s_wait_asynccnt(0);
    }
#else
    {
        const uint4* src = (const uint4*)(A + (size_t)blockIdx.y * 16 * K);
        uint4*       dst = (uint4*)sA;
        const int nvec = (16 * K) >> 3;
        for (int i = tid; i < nvec; i += 256) dst[i] = src[i];
    }
#endif
    __syncthreads();

    v8f acc0 = {}; v8f acc1 = {};
    const __bf16* w0   = W  + (size_t)(n0 + mrow)      * K + khalf * 16;
    const __bf16* w1   = W  + (size_t)(n0 + 16 + mrow) * K + khalf * 16;
    const __bf16* arow = sA + mrow * K + khalf * 16;

    for (int k0 = 0; k0 < K; k0 += 32) {
        BF16x16 at;
        at.lo = *(const uint4*)(arow + k0);        // A[m, k0 + khalf*16 + 0..7]
        at.hi = *(const uint4*)(arow + k0 + 8);    // A[m, k0 + khalf*16 + 8..15]
        v16bf a  = __builtin_bit_cast(v16bf, at);
        v16bf b0 = *(const v16bf*)(w0 + k0);       // 32B: W[n0+mrow, k0+khalf*16 ..]
        v16bf b1 = *(const v16bf*)(w1 + k0);
        __builtin_prefetch(w0 + k0 + 64, 0, 1);    // global_prefetch next K slab
        acc0 = __builtin_amdgcn_wmma_f32_16x16x32_bf16(false, a, false, b0,
                                                       (short)0, acc0, false, false);
        acc1 = __builtin_amdgcn_wmma_f32_16x16x32_bf16(false, a, false, b1,
                                                       (short)0, acc1, false, false);
    }

    const float bv0 = bias ? bias[n0 + mrow]      : 0.0f;
    const float bv1 = bias ? bias[n0 + 16 + mrow] : 0.0f;
    // f32 C/D layout: VGPR r -> M = r + khalf*8, N = lane&15
    float* crow = C + (size_t)(blockIdx.y * 16 + khalf * 8) * N + n0 + mrow;
#pragma unroll
    for (int r = 0; r < 8; ++r) {
        crow[(size_t)r * N]      = acc0[r] + bv0;
        crow[(size_t)r * N + 16] = acc1[r] + bv1;
    }
}

// ---------------------------------------------------------------------------
// fp32 -> bf16 convert with destination stride/offset (also builds [Wih|Whh])
// ---------------------------------------------------------------------------
__global__ void k_convert(const float* __restrict__ src, __bf16* __restrict__ dst,
                          int rows, int cols, int dld, int doff)
{
    size_t i = (size_t)blockIdx.x * 256 + threadIdx.x;
    if (i >= (size_t)rows * cols) return;
    int r = (int)(i / cols), c = (int)(i % cols);
    dst[(size_t)r * dld + doff + c] = (__bf16)src[i];
}

__global__ void k_zero(float* p, int n)
{
    int i = blockIdx.x * 256 + threadIdx.x;
    if (i < n) p[i] = 0.0f;
}

// src embedding gather -> bf16 (2048 x 512)
__global__ void k_embed_src(const int* __restrict__ x, const float* __restrict__ emb,
                            __bf16* __restrict__ dst)
{
    size_t i = (size_t)blockIdx.x * 256 + threadIdx.x;    // 2048*512
    if (i >= (size_t)2048 * 512) return;
    int row = (int)(i >> 9), c = (int)(i & 511);
    dst[i] = (__bf16)emb[(size_t)x[row] * 512 + c];
}

// tgt embedding gather for step t -> left half of Acat0 (16 x 2048, cols 0..1023)
__global__ void k_embed_tgt(const int* __restrict__ y, int t,
                            const float* __restrict__ emb, __bf16* __restrict__ dst)
{
    int i = blockIdx.x * 256 + threadIdx.x;               // 16*1024
    if (i >= 16 * 1024) return;
    int b = i >> 10, c = i & 1023;
    dst[(size_t)b * 2048 + c] = (__bf16)emb[(size_t)y[b * 128 + t] * 1024 + c];
}

// fp32 h-state -> bf16 at [b*ld + off + c]
__global__ void k_h2bf(const float* __restrict__ h, __bf16* __restrict__ dst,
                       int ld, int off)
{
    int i = blockIdx.x * 256 + threadIdx.x;               // 16*1024
    if (i >= 16 * 1024) return;
    int b = i >> 10, c = i & 1023;
    dst[(size_t)b * ld + off + c] = (__bf16)h[i];
}

// encoder zero-step activation: gates g (2048 x 2048), Dh=512
// h = sigmoid(o) * tanh( sigmoid(i) * tanh(gg) )   (f gate unused at c0=0)
__global__ void k_enc_act(const float* __restrict__ g, float* __restrict__ hF,
                          __bf16* __restrict__ hB, int colOff)
{
    size_t i = (size_t)blockIdx.x * 256 + threadIdx.x;    // 2048*512
    if (i >= (size_t)2048 * 512) return;
    int m = (int)(i >> 9), c = (int)(i & 511);
    const float* gr = g + (size_t)m * 2048;
    float iv = gr[c], gg = gr[1024 + c], ov = gr[1536 + c];
    float cc = sigmoidf(iv) * tanhf(gg);
    float h  = sigmoidf(ov) * tanhf(cc);
    hF[(size_t)m * 1024 + colOff + c] = h;
    if (hB) hB[(size_t)m * 1024 + colOff + c] = (__bf16)h;
}

// decoder LSTM activation: gates g (16 x 4096), cell_mem (16 x 1024)
// c = sig(f)*cell_mem + sig(i)*tanh(gg);  h = sig(o)*tanh(c)
__global__ void k_dec_act(const float* __restrict__ g, const float* __restrict__ cm,
                          float* __restrict__ hnew, __bf16* __restrict__ bdst, int bld)
{
    int i = blockIdx.x * 256 + threadIdx.x;               // 16*1024
    if (i >= 16 * 1024) return;
    int b = i >> 10, c = i & 1023;
    const float* gr = g + (size_t)b * 4096;
    float iv = gr[c], fv = gr[1024 + c], gg = gr[2048 + c], ov = gr[3072 + c];
    float cc = sigmoidf(fv) * cm[i] + sigmoidf(iv) * tanhf(gg);
    float h  = sigmoidf(ov) * tanhf(cc);
    hnew[i] = h;
    bdst[(size_t)b * bld + c] = (__bf16)h;
}

// attention for one (layer n, batch b): scores over L=128, softmax, context
// grid = 32 (n*16+b), block = 128 (one thread per source position)
__global__ void k_attn(const float* __restrict__ h0f, const float* __restrict__ h1f,
                       const float* __restrict__ hstate,
                       const float* __restrict__ attn_W, const float* __restrict__ attn_b,
                       float* __restrict__ cellmem)
{
    const int n = blockIdx.x >> 4, b = blockIdx.x & 15;
    const float* enc = (n == 0 ? h0f : h1f) + (size_t)b * 128 * 1024;
    const float* awq = attn_W;
    const float* awk = attn_W + 1024;
    const float* h   = hstate + (size_t)n * 16 * 1024 + (size_t)b * 1024;
    const int l = threadIdx.x;
    __shared__ float red[128];
    __shared__ float aw[128];

    // q = <h, aw_q> (cooperative) + attn_b
    float qp = 0.0f;
    for (int d = l; d < 1024; d += 128) qp += h[d] * awq[d];
    red[l] = qp; __syncthreads();
    for (int s = 64; s > 0; s >>= 1) { if (l < s) red[l] += red[l + s]; __syncthreads(); }
    float q = red[0] + attn_b[0]; __syncthreads();

    // score_l = q + <enc[l], aw_k>
    const float* er = enc + (size_t)l * 1024;
    float dot = 0.0f;
    for (int d = 0; d < 1024; ++d) dot += er[d] * awk[d];
    float sc = q + dot;

    red[l] = sc; __syncthreads();
    for (int s = 64; s > 0; s >>= 1) { if (l < s) red[l] = fmaxf(red[l], red[l + s]); __syncthreads(); }
    float mx = red[0]; __syncthreads();
    float e = __expf(sc - mx);
    red[l] = e; __syncthreads();
    for (int s = 64; s > 0; s >>= 1) { if (l < s) red[l] += red[l + s]; __syncthreads(); }
    float inv = 1.0f / red[0]; __syncthreads();
    aw[l] = e * inv; __syncthreads();

    // context: each thread owns 8 dims
    float acc[8] = {};
    for (int l2 = 0; l2 < 128; ++l2) {
        float a = aw[l2];
        const float* e2 = enc + (size_t)l2 * 1024 + l * 8;
#pragma unroll
        for (int j = 0; j < 8; ++j) acc[j] += a * e2[j];
    }
    float* cmo = cellmem + (size_t)n * 16 * 1024 + (size_t)b * 1024 + l * 8;
#pragma unroll
    for (int j = 0; j < 8; ++j) cmo[j] = acc[j];
}

// per-row log-softmax NLL for step t (grid = 16 rows, block = 256)
__global__ void k_nll(const float* __restrict__ logits, const int* __restrict__ y,
                      int t, float* __restrict__ nll)
{
    const int b = blockIdx.x, tid = threadIdx.x;
    const float* L = logits + (size_t)b * 32000;
    __shared__ float red[256];
    float mx = -3.4e38f;
    for (int i = tid; i < 32000; i += 256) mx = fmaxf(mx, L[i]);
    red[tid] = mx; __syncthreads();
    for (int s = 128; s > 0; s >>= 1) { if (tid < s) red[tid] = fmaxf(red[tid], red[tid + s]); __syncthreads(); }
    mx = red[0]; __syncthreads();
    float sm = 0.0f;
    for (int i = tid; i < 32000; i += 256) sm += __expf(L[i] - mx);
    red[tid] = sm; __syncthreads();
    for (int s = 128; s > 0; s >>= 1) { if (tid < s) red[tid] += red[tid + s]; __syncthreads(); }
    if (tid == 0) {
        int tgt = y[b * 128 + t];
        nll[t * 16 + b] = (mx + __logf(red[0])) - L[tgt];
    }
}

// deterministic final reduction: out = sum(nll*mask)/max(sum(mask),1)
__global__ void k_reduce(const float* __restrict__ nll, const int* __restrict__ y,
                         float* __restrict__ out)
{
    __shared__ float rs[256], rm[256];
    int tid = threadIdx.x;
    float s = 0.0f, m = 0.0f;
    for (int i = tid; i < 2048; i += 256) {
        int t = i / 16, b = i % 16;
        float msk = (y[b * 128 + t] != 0) ? 1.0f : 0.0f;
        s += nll[i] * msk; m += msk;
    }
    rs[tid] = s; rm[tid] = m; __syncthreads();
    for (int st = 128; st > 0; st >>= 1) {
        if (tid < st) { rs[tid] += rs[tid + st]; rm[tid] += rm[tid + st]; }
        __syncthreads();
    }
    if (tid == 0) out[0] = rs[0] / fmaxf(rm[0], 1.0f);
}

// ---------------------------------------------------------------------------
extern "C" void kernel_launch(void* const* d_in, const int* in_sizes, int n_in,
                              void* d_out, int out_size, void* d_ws, size_t ws_size,
                              hipStream_t stream)
{
    (void)in_sizes; (void)n_in; (void)out_size; (void)ws_size;
    const int*   x         = (const int*)  d_in[0];
    const int*   y         = (const int*)  d_in[1];
    const float* src_emb   = (const float*)d_in[2];
    const float* tgt_emb   = (const float*)d_in[3];
    const float* enc_Wih_0 = (const float*)d_in[4];
    const float* enc_b_0   = (const float*)d_in[5];
    const float* enc_Wih_1 = (const float*)d_in[6];
    const float* enc_b_1   = (const float*)d_in[7];
    const float* dec_Wih   = (const float*)d_in[8];
    const float* dec_Whh   = (const float*)d_in[9];
    const float* dec_b     = (const float*)d_in[10];
    const float* attn_W    = (const float*)d_in[11];
    const float* attn_b    = (const float*)d_in[12];
    const float* head_W    = (const float*)d_in[13];
    const float* head_b    = (const float*)d_in[14];
    float*       out       = (float*)d_out;

    // workspace carve-up (256B aligned)
    char*  w   = (char*)d_ws;
    size_t off = 0;
    auto alloc = [&](size_t bytes) -> void* {
        void* p = w + off;
        off = (off + bytes + 255) & ~(size_t)255;
        return p;
    };
    __bf16* ex_bf   = (__bf16*)alloc((size_t)2048 * 512 * 2);
    __bf16* encW0   = (__bf16*)alloc((size_t)4096 * 512 * 2);       // 2 x (2048x512)
    __bf16* encW1   = (__bf16*)alloc((size_t)4096 * 1024 * 2);      // 2 x (2048x1024)
    __bf16* wcat    = (__bf16*)alloc((size_t)2 * 4096 * 2048 * 2);  // [Wih|Whh] per layer
    __bf16* headWb  = (__bf16*)alloc((size_t)32000 * 1024 * 2);
    float*  gbuf    = (float*) alloc((size_t)2048 * 2048 * 4);      // gates scratch
    float*  h0f     = (float*) alloc((size_t)2048 * 1024 * 4);      // enc_hs[0]
    float*  h1f     = (float*) alloc((size_t)2048 * 1024 * 4);      // enc_hs[1]
    __bf16* h0bf    = (__bf16*)alloc((size_t)2048 * 1024 * 2);
    float*  hstate  = (float*) alloc((size_t)2 * 16 * 1024 * 4);
    float*  cellmem = (float*) alloc((size_t)2 * 16 * 1024 * 4);
    __bf16* acat0   = (__bf16*)alloc((size_t)16 * 2048 * 2);        // [emb | h0]
    __bf16* acat1   = (__bf16*)alloc((size_t)16 * 2048 * 2);        // [h0' | h1]
    __bf16* hheadbf = (__bf16*)alloc((size_t)16 * 1024 * 2);
    float*  logits  = (float*) alloc((size_t)16 * 32000 * 4);
    float*  nllbuf  = (float*) alloc((size_t)2048 * 4);

    auto cgrid = [](size_t n) { return (unsigned)((n + 255) / 256); };

    // ---- one-time (per launch) weight conversions -------------------------
    k_convert<<<cgrid((size_t)4096 * 512), 256, 0, stream>>>(enc_Wih_0, encW0, 4096, 512, 512, 0);
    k_convert<<<cgrid((size_t)4096 * 1024), 256, 0, stream>>>(enc_Wih_1, encW1, 4096, 1024, 1024, 0);
    for (int j = 0; j < 2; ++j) {
        k_convert<<<cgrid((size_t)4096 * 1024), 256, 0, stream>>>(
            dec_Wih + (size_t)j * 4096 * 1024, wcat + (size_t)j * 4096 * 2048, 4096, 1024, 2048, 0);
        k_convert<<<cgrid((size_t)4096 * 1024), 256, 0, stream>>>(
            dec_Whh + (size_t)j * 4096 * 1024, wcat + (size_t)j * 4096 * 2048, 4096, 1024, 2048, 1024);
    }
    k_convert<<<cgrid((size_t)32000 * 1024), 256, 0, stream>>>(head_W, headWb, 32000, 1024, 1024, 0);
    k_embed_src<<<cgrid((size_t)2048 * 512), 256, 0, stream>>>(x, src_emb, ex_bf);
    k_zero<<<cgrid(2 * 16 * 1024), 256, 0, stream>>>(hstate, 2 * 16 * 1024);

    // ---- encoder ----------------------------------------------------------
    dim3 ge(2048 / 256, 2048 / 16);                     // (8, 128)
    for (int d = 0; d < 2; ++d) {
        k_gemm_bf16<<<ge, 256, 16 * 512 * 2, stream>>>(
            ex_bf, encW0 + (size_t)d * 2048 * 512, enc_b_0 + (size_t)d * 2048, gbuf, 2048, 512);
        k_enc_act<<<cgrid((size_t)2048 * 512), 256, 0, stream>>>(gbuf, h0f, h0bf, d * 512);
    }
    for (int d = 0; d < 2; ++d) {
        k_gemm_bf16<<<ge, 256, 16 * 1024 * 2, stream>>>(
            h0bf, encW1 + (size_t)d * 2048 * 1024, enc_b_1 + (size_t)d * 2048, gbuf, 2048, 1024);
        k_enc_act<<<cgrid((size_t)2048 * 512), 256, 0, stream>>>(gbuf, h1f, (__bf16*)nullptr, d * 512);
    }

    // ---- decoder scan (sequential over 128 target steps) ------------------
    dim3 gd(4096 / 256, 1);                             // gate GEMMs, M=16
    dim3 gh(32000 / 256, 1);                            // head GEMM, M=16
    for (int t = 0; t < 128; ++t) {
        k_embed_tgt<<<64, 256, 0, stream>>>(y, t, tgt_emb, acat0);          // emb -> Acat0[:, :1024]
        k_h2bf<<<64, 256, 0, stream>>>(hstate,          acat0, 2048, 1024); // old h0 -> Acat0 right
        k_h2bf<<<64, 256, 0, stream>>>(hstate + 16384,  acat1, 2048, 1024); // old h1 -> Acat1 right
        k_attn<<<32, 128, 0, stream>>>(h0f, h1f, hstate, attn_W, attn_b, cellmem);

        k_gemm_bf16<<<gd, 256, 16 * 2048 * 2, stream>>>(acat0, wcat, dec_b, gbuf, 4096, 2048);
        k_dec_act<<<64, 256, 0, stream>>>(gbuf, cellmem, hstate, acat1, 2048);  // h0' -> Acat1 left

        k_gemm_bf16<<<gd, 256, 16 * 2048 * 2, stream>>>(
            acat1, wcat + (size_t)4096 * 2048, dec_b + 4096, gbuf, 4096, 2048);
        k_dec_act<<<64, 256, 0, stream>>>(gbuf, cellmem + 16384, hstate + 16384, hheadbf, 1024);

        k_gemm_bf16<<<gh, 256, 16 * 1024 * 2, stream>>>(hheadbf, headWb, head_b, logits, 32000, 1024);
        k_nll<<<16, 256, 0, stream>>>(logits, y, t, nllbuf);
    }
    k_reduce<<<1, 256, 0, stream>>>(nllbuf, y, out);
}
